// BiLSTM_CRF_84628035601020
// MI455X (gfx1250) — matrix operand order repaired and checked by reading, not compiled, not run on hardware
//
#include <hip/hip_runtime.h>

// ---------------------------------------------------------------------------
// BiLSTM-CRF for MI455X (gfx1250, wave32, WMMA)
//
// Phases (all on `stream`, sequential):
//   1) gather+convert: emb[words] -> bf16 x, W_ih/W_hh fp32 -> bf16 (ws)
//   2) input GEMM (WMMA bf16, f32 accum): pre = x @ W_ih^T + b   [T,4H] x2 dirs
//   3) persistent recurrent kernel: 32 WGs/dir, W_hh slice (256KB) pinned in
//      LDS (320KB/WGP), global monotonic-counter barrier per timestep,
//      double-buffered h (L2-resident), b128 LDS reads in the inner dot
//   4) output projection -> frames [T,18]
//   5) single-wave Viterbi (logsumexp alpha scan + argmax backpointers + BT)
// ---------------------------------------------------------------------------

#define T_LEN   4096
#define E_DIM   512
#define H_DIM   1024
#define G4H     4096      // 4*H
#define NTAG    18
#define BEG_TAG 16
#define END_TAG 17
#define NEG_INF (-10000.0f)

#define NWG_DIR     32
#define REC_THREADS 256
#define WROW_STRIDE 1032  // 1024 + 8 bf16 pad (16B) to break LDS bank alignment

typedef __bf16 bf16;
typedef __attribute__((ext_vector_type(8)))  __bf16 v8bf;
typedef __attribute__((ext_vector_type(16))) __bf16 v16bf;
typedef __attribute__((ext_vector_type(8)))  float  v8f;

// ---------------- workspace layout (bytes) ----------------
static const size_t OFF_XBF  = 0;                        // T*E*2        = 4 MiB
static const size_t OFF_WIHF = OFF_XBF  + 4194304;       // 4H*E*2       = 4 MiB
static const size_t OFF_WIHB = OFF_WIHF + 4194304;
static const size_t OFF_WHHF = OFF_WIHB + 4194304;       // 4H*H*2       = 8 MiB
static const size_t OFF_WHHB = OFF_WHHF + 8388608;
static const size_t OFF_PREF = OFF_WHHB + 8388608;       // T*4H*4       = 64 MiB
static const size_t OFF_PREB = OFF_PREF + 67108864;
static const size_t OFF_HBUF = OFF_PREB + 67108864;      // 2*2*H*4
static const size_t OFF_HSF  = OFF_HBUF + 16384;         // T*H*4        = 16 MiB
static const size_t OFF_HSB  = OFF_HSF  + 16777216;
static const size_t OFF_FRM  = OFF_HSB  + 16777216;      // T*18*4
static const size_t OFF_BPS  = OFF_FRM  + 294912;        // T*18*4 (int)
static const size_t OFF_CNT  = OFF_BPS  + 294912;        // barrier counters

// ---------------- small helpers ----------------
__global__ void k_init_counters(unsigned* c) {
  if (threadIdx.x < 64) c[threadIdx.x] = 0u;
}

__global__ void k_f32_to_bf16(const float* __restrict__ src,
                              bf16* __restrict__ dst, int n) {
  int i = blockIdx.x * blockDim.x + threadIdx.x;
  int stride = gridDim.x * blockDim.x;
  for (; i < n; i += stride) dst[i] = (bf16)src[i];
}

__global__ void k_gather_embed(const int* __restrict__ words,
                               const float* __restrict__ emb,
                               bf16* __restrict__ xbf) {
  int i = blockIdx.x * blockDim.x + threadIdx.x;     // over T*E
  if (i >= T_LEN * E_DIM) return;
  int t = i / E_DIM, e = i - t * E_DIM;
  xbf[i] = (bf16)emb[(size_t)words[t] * E_DIM + e];
}

// ---------------- WMMA input GEMM: pre = x @ W^T + b ----------------
// A = x [T,E] bf16 row-major (K contiguous), B = W [4H,E] bf16 row-major,
// i.e. B's N index is the row -> each B lane reads 16 contiguous K. NT-GEMM:
// no transpose loads needed. Per-wave output: 16(M) x 64(N), K-loop of 16.
__device__ inline v16bf load_k16(const bf16* p) {
  v8bf lo = *(const v8bf*)p;
  v8bf hi = *(const v8bf*)(p + 8);
  v16bf r;
#pragma unroll
  for (int e = 0; e < 8; ++e) { r[e] = lo[e]; r[e + 8] = hi[e]; }
  return r;
}

__global__ __launch_bounds__(256)
void k_input_gemm(const bf16* __restrict__ xbf,
                  const bf16* __restrict__ wih,   // [4096,512]
                  const float* __restrict__ bias, // [4096]
                  float* __restrict__ pre) {      // [T,4096]
  const int lane = threadIdx.x & 31;
  const int wave = threadIdx.x >> 5;
  const int m0 = blockIdx.y * 128 + wave * 16;
  const int n0 = blockIdx.x * 64;
  const int mrow = m0 + (lane & 15);
  const int ka = (lane >> 4) * 16;   // A: one 16-elem K chunk, halves split K
  const int kb = (lane >> 4) * 16;   // B: one 16-elem K chunk, halves split K

  v8f acc0 = {}, acc1 = {}, acc2 = {}, acc3 = {};
  const bf16* arow  = xbf + (size_t)mrow * E_DIM;
  const bf16* brow0 = wih + (size_t)(n0 +  0 + (lane & 15)) * E_DIM;
  const bf16* brow1 = wih + (size_t)(n0 + 16 + (lane & 15)) * E_DIM;
  const bf16* brow2 = wih + (size_t)(n0 + 32 + (lane & 15)) * E_DIM;
  const bf16* brow3 = wih + (size_t)(n0 + 48 + (lane & 15)) * E_DIM;

  for (int k0 = 0; k0 < E_DIM; k0 += 32) {
    v8bf alo = *(const v8bf*)(arow + k0 + ka);
    v8bf ahi = *(const v8bf*)(arow + k0 + ka + 8);
    v16bf A;
#pragma unroll
    for (int e = 0; e < 8; ++e) { A[e] = alo[e]; A[e + 8] = ahi[e]; }
    if (k0 + 32 < E_DIM) {        // WGP-scope prefetch of next K slab
      __builtin_prefetch(arow + k0 + 32, 0, 3);
      __builtin_prefetch(brow0 + k0 + 32, 0, 3);
    }
    v16bf B0 = load_k16(brow0 + k0 + kb);
    v16bf B1 = load_k16(brow1 + k0 + kb);
    v16bf B2 = load_k16(brow2 + k0 + kb);
    v16bf B3 = load_k16(brow3 + k0 + kb);
    acc0 = __builtin_amdgcn_wmma_f32_16x16x32_bf16(false, A, false, B0,
                                                   (short)0, acc0, false, false);
    acc1 = __builtin_amdgcn_wmma_f32_16x16x32_bf16(false, A, false, B1,
                                                   (short)0, acc1, false, false);
    acc2 = __builtin_amdgcn_wmma_f32_16x16x32_bf16(false, A, false, B2,
                                                   (short)0, acc2, false, false);
    acc3 = __builtin_amdgcn_wmma_f32_16x16x32_bf16(false, A, false, B3,
                                                   (short)0, acc3, false, false);
  }

  // C/D layout: VGPR r -> M = r (lanes 0-15) / r+8 (lanes 16-31); N = lane&15
  const int col = lane & 15;
  const int rbase = (lane >> 4) * 8;
  const float bi0 = bias[n0 +  0 + col];
  const float bi1 = bias[n0 + 16 + col];
  const float bi2 = bias[n0 + 32 + col];
  const float bi3 = bias[n0 + 48 + col];
#pragma unroll
  for (int r = 0; r < 8; ++r) {
    size_t row = (size_t)(m0 + rbase + r) * G4H;
    pre[row + n0 +  0 + col] = acc0[r] + bi0;
    pre[row + n0 + 16 + col] = acc1[r] + bi1;
    pre[row + n0 + 32 + col] = acc2[r] + bi2;
    pre[row + n0 + 48 + col] = acc3[r] + bi3;
  }
}

// ---------------- persistent recurrent kernel ----------------
// 64 WGs total: dir = blk>>5 (0 fwd, 1 bwd), w = blk&31 owns h-slice
// n in [w*32, w*32+32). LDS holds the 4 gate rows for those n: 4*32*1024 bf16.
__global__ __launch_bounds__(REC_THREADS)
void k_recurrent(const bf16* __restrict__ whh_f, const bf16* __restrict__ whh_b,
                 const float* __restrict__ pre_f, const float* __restrict__ pre_b,
                 const float* __restrict__ h0, const float* __restrict__ c0,
                 float* __restrict__ hbuf,       // [2 dirs][2 bufs][H]
                 float* __restrict__ hs_f, float* __restrict__ hs_b,
                 unsigned* __restrict__ counters) {
  extern __shared__ char smem[];
  bf16*  lds_w = (bf16*)smem;                          // 128 * WROW_STRIDE
  float* h_sh  = (float*)(smem + 128 * WROW_STRIDE * 2);
  float* red   = h_sh + H_DIM;                         // [256]
  float* zsh   = red + REC_THREADS;                    // [128]

  const int dir = blockIdx.x >> 5;
  const int w   = blockIdx.x & 31;
  const int tid = threadIdx.x;
  const bf16*  whh = dir ? whh_b : whh_f;
  const float* pre = dir ? pre_b : pre_f;
  float* hs  = dir ? hs_b : hs_f;
  float* hb0 = hbuf + (size_t)dir * 2 * H_DIM;
  unsigned* cnt = counters + dir * 32;                 // separate cachelines

  // Stage W_hh slice into LDS: 2 threads per (gate,row), b128 copies
  {
    const int row  = tid >> 1;                         // 0..127 = gate*32+i
    const int half = tid & 1;
    const int g2 = row >> 5, i2 = row & 31;
    const uint4* s4 = (const uint4*)(whh +
        ((size_t)(g2 * H_DIM + w * 32 + i2)) * H_DIM + half * 512);
    uint4* d4 = (uint4*)(lds_w + row * WROW_STRIDE + half * 512);
#pragma unroll 4
    for (int q = 0; q < 64; ++q) d4[q] = s4[q];
  }

  float c_reg = 0.f;
  if (tid < 32) {
    int n = w * 32 + tid;
    c_reg = c0[dir * H_DIM + n];
    hb0[n] = h0[dir * H_DIM + n];                      // buffer 0 = h0
  }
  __syncthreads();
  __threadfence();
  if (tid == 0) atomicAdd(cnt, 1u);                    // init arrival

  const int o  = tid & 127;                            // output: gate*32+i
  const int kh = tid >> 7;                             // K half
  const int gg = o >> 5, il = o & 31;
  volatile unsigned* vcnt = cnt;
  const v8bf* wv = (const v8bf*)(lds_w + o * WROW_STRIDE + kh * 512);

  for (int step = 0; step < T_LEN; ++step) {
    if (tid == 0) {
      unsigned target = (unsigned)(step + 1) * NWG_DIR;
      while (*vcnt < target) __builtin_amdgcn_s_sleep(2);
      __threadfence();
    }
    __syncthreads();

    const int t = dir ? (T_LEN - 1 - step) : step;
    const float* hin = hb0 + (step & 1) * H_DIM;
    for (int k = tid; k < H_DIM; k += REC_THREADS) h_sh[k] = hin[k];
    __syncthreads();

    // z_o = pre[t,o] + dot(h, W_hh[row(o),:])   split-K over 2 threads.
    // Weight side: 16B LDS reads (ds_load_b128), 8-wide cvt+fma body.
    float acc = 0.f;
    const float* hk = h_sh + kh * 512;
#pragma unroll 4
    for (int q = 0; q < 64; ++q) {
      v8bf wq = wv[q];
      const float* hq = hk + q * 8;
#pragma unroll
      for (int e = 0; e < 8; ++e) acc += hq[e] * (float)wq[e];
    }
    red[tid] = acc;
    __syncthreads();
    if (tid < 128) {
      zsh[tid] = red[tid] + red[tid + 128] +
                 pre[(size_t)t * G4H + gg * H_DIM + w * 32 + il];
    }
    __syncthreads();
    if (tid < 32) {
      int n = w * 32 + tid;
      float iz = zsh[tid], fz = zsh[32 + tid];
      float gz = zsh[64 + tid], oz = zsh[96 + tid];
      float ig = 1.f / (1.f + __expf(-iz));
      float fg = 1.f / (1.f + __expf(-fz));
      float og = 1.f / (1.f + __expf(-oz));
      c_reg = fg * c_reg + ig * tanhf(gz);
      float hnew = og * tanhf(c_reg);
      hb0[((step + 1) & 1) * H_DIM + n] = hnew;
      hs[(size_t)t * H_DIM + n] = hnew;
    }
    __threadfence();
    __syncthreads();
    if (tid == 0) atomicAdd(cnt, 1u);
  }
}

// ---------------- output projection ----------------
__global__ void k_frames(const float* __restrict__ hs_f,
                         const float* __restrict__ hs_b,
                         const float* __restrict__ w_out,  // [2H,18]
                         const float* __restrict__ b_out,
                         float* __restrict__ frames) {     // [T,18]
  int idx = blockIdx.x * blockDim.x + threadIdx.x;
  if (idx >= T_LEN * NTAG) return;
  int t = idx / NTAG, tag = idx - t * NTAG;
  float acc = b_out[tag];
  const float* hf = hs_f + (size_t)t * H_DIM;
  const float* hb = hs_b + (size_t)t * H_DIM;
  for (int h = 0; h < H_DIM; ++h) {
    acc += hf[h] * w_out[h * NTAG + tag];
    acc += hb[h] * w_out[(H_DIM + h) * NTAG + tag];
  }
  frames[idx] = acc;
}

// ---------------- Viterbi (single wave32) ----------------
__global__ void k_viterbi(const float* __restrict__ frames,
                          const float* __restrict__ trans,  // [18,18]
                          float* __restrict__ out,          // [1 + T]
                          int* __restrict__ bps) {          // [T,18]
  __shared__ float alpha[NTAG], nalpha[NTAG];
  const int j = threadIdx.x;
  if (j < NTAG) alpha[j] = (j == BEG_TAG) ? 0.f : NEG_INF;
  __syncthreads();

  for (int t = 0; t < T_LEN; ++t) {
    if (j < NTAG) {
      float v[NTAG];
      float m = -3.0e38f; int am = 0;
      for (int i = 0; i < NTAG; ++i) {
        v[i] = alpha[i] + trans[i * NTAG + j];
        if (v[i] > m) { m = v[i]; am = i; }
      }
      float s = 0.f;
      for (int i = 0; i < NTAG; ++i) s += __expf(v[i] - m);
      nalpha[j] = frames[t * NTAG + j] + m + __logf(s);
      bps[t * NTAG + j] = am;
    }
    __syncthreads();
    if (j < NTAG) alpha[j] = nalpha[j];
    __syncthreads();
  }

  if (j == 0) {
    float fin[NTAG];
    float m = -3.0e38f; int bl = 0;
    for (int i = 0; i < NTAG; ++i) {
      fin[i] = alpha[i] + trans[i * NTAG + END_TAG];
      if (fin[i] > m) { m = fin[i]; bl = i; }
    }
    float s = 0.f;
    for (int i = 0; i < NTAG; ++i) s += __expf(fin[i] - m);
    out[0] = m + __logf(s);                       // score (logsumexp)
    int tag = bl;
    out[1 + (T_LEN - 1)] = (float)tag;
    for (int t = T_LEN - 1; t >= 1; --t) {        // path[t-1] = bps[t][path[t]]
      tag = bps[t * NTAG + tag];
      out[1 + (t - 1)] = (float)tag;
    }
  }
}

// ---------------- host launcher ----------------
extern "C" void kernel_launch(void* const* d_in, const int* in_sizes, int n_in,
                              void* d_out, int out_size, void* d_ws, size_t ws_size,
                              hipStream_t stream) {
  const int*   words = (const int*)d_in[0];
  const float* h0    = (const float*)d_in[1];
  const float* c0    = (const float*)d_in[2];
  const float* emb   = (const float*)d_in[3];
  const float* wihf  = (const float*)d_in[4];
  const float* whhf  = (const float*)d_in[5];
  const float* bf_   = (const float*)d_in[6];
  const float* wihb  = (const float*)d_in[7];
  const float* whhb  = (const float*)d_in[8];
  const float* bb_   = (const float*)d_in[9];
  const float* wout  = (const float*)d_in[10];
  const float* bout  = (const float*)d_in[11];
  const float* trans = (const float*)d_in[12];

  char* ws = (char*)d_ws;
  bf16*  xbf     = (bf16*)(ws + OFF_XBF);
  bf16*  wihf_bf = (bf16*)(ws + OFF_WIHF);
  bf16*  wihb_bf = (bf16*)(ws + OFF_WIHB);
  bf16*  whhf_bf = (bf16*)(ws + OFF_WHHF);
  bf16*  whhb_bf = (bf16*)(ws + OFF_WHHB);
  float* pre_f   = (float*)(ws + OFF_PREF);
  float* pre_b   = (float*)(ws + OFF_PREB);
  float* hbuf    = (float*)(ws + OFF_HBUF);
  float* hs_f    = (float*)(ws + OFF_HSF);
  float* hs_b    = (float*)(ws + OFF_HSB);
  float* frames  = (float*)(ws + OFF_FRM);
  int*   bps     = (int*)(ws + OFF_BPS);
  unsigned* cnts = (unsigned*)(ws + OFF_CNT);

  k_init_counters<<<1, 64, 0, stream>>>(cnts);
  k_gather_embed<<<(T_LEN * E_DIM + 255) / 256, 256, 0, stream>>>(words, emb, xbf);
  k_f32_to_bf16<<<2048, 256, 0, stream>>>(wihf, wihf_bf, G4H * E_DIM);
  k_f32_to_bf16<<<2048, 256, 0, stream>>>(wihb, wihb_bf, G4H * E_DIM);
  k_f32_to_bf16<<<2048, 256, 0, stream>>>(whhf, whhf_bf, G4H * H_DIM);
  k_f32_to_bf16<<<2048, 256, 0, stream>>>(whhb, whhb_bf, G4H * H_DIM);

  dim3 ggrid(G4H / 64, T_LEN / 128);          // (N tiles, M tiles)
  k_input_gemm<<<ggrid, 256, 0, stream>>>(xbf, wihf_bf, bf_, pre_f);
  k_input_gemm<<<ggrid, 256, 0, stream>>>(xbf, wihb_bf, bb_, pre_b);

  size_t rec_smem = (size_t)128 * WROW_STRIDE * 2         // W slice (bf16)
                  + (H_DIM + REC_THREADS + 128) * sizeof(float);
  k_recurrent<<<2 * NWG_DIR, REC_THREADS, rec_smem, stream>>>(
      whhf_bf, whhb_bf, pre_f, pre_b, h0, c0, hbuf, hs_f, hs_b, cnts);

  k_frames<<<(T_LEN * NTAG + 255) / 256, 256, 0, stream>>>(
      hs_f, hs_b, wout, bout, frames);
  k_viterbi<<<1, 32, 0, stream>>>(frames, trans, (float*)d_out, bps);
}
